// A5ExactScan_62534723830141
// MI455X (gfx1250) — 8-bit matrix operand order repaired and checked by reading, compiled, not run on hardware
//
#include <hip/hip_runtime.h>
#include <stdint.h>

typedef __attribute__((ext_vector_type(8))) int v8i;
typedef __attribute__((ext_vector_type(4))) int v4i;
typedef __attribute__((ext_vector_type(4))) unsigned int v4u;

constexpr int kB = 4096;
constexpr int kT = 4096;
constexpr int kN = 60;
constexpr int kC = 16;
constexpr int kL = kT / kC;

__global__ __launch_bounds__(64) void zscan_chunk_perm(
    const int* __restrict__ ids, const int* __restrict__ mul,
    uint8_t* __restrict__ perms)
{
  __shared__ uint8_t mulb[kN * kN + 16];
  __shared__ int tok[kL];

  const int tid = threadIdx.x;
  const int blk = blockIdx.x;
  const int b = blk / kC;
  const int c = blk % kC;
  const int* src = ids + (size_t)b * kT + (size_t)c * kL;

  __builtin_prefetch(src + tid * 4, 0, 1);

#if defined(__HIP_DEVICE_COMPILE__) && \
    __has_builtin(__builtin_amdgcn_tensor_load_to_lds) && \
    __has_builtin(__builtin_amdgcn_s_wait_tensorcnt)
  if (tid == 0) {
    const uint64_t gaddr = (uint64_t)(uintptr_t)src;
    const uint32_t laddr = (uint32_t)(uintptr_t)&tok[0];
    v4u g0;
    g0.x = 1u;
    g0.y = laddr;
    g0.z = (uint32_t)(gaddr & 0xFFFFFFFFull);
    g0.w = (uint32_t)((gaddr >> 32) & 0x01FFFFFFull) | (2u << 30);
    v8i g1;
    g1[0] = (int)(2u << 16);
    g1[1] = (int)((uint32_t)(kL & 0xFFFF) << 16);
    g1[2] = (int)(1u << 16);
    g1[3] = (int)((uint32_t)(kL & 0xFFFF) << 16);
    g1[4] = 1;
    g1[5] = kL;
    g1[6] = (int)((uint32_t)(kL & 0xFFFF) << 16);
    g1[7] = 0;
    v4i gz4 = {0, 0, 0, 0};
    v8i gz8 = {0, 0, 0, 0, 0, 0, 0, 0};
    __builtin_amdgcn_tensor_load_to_lds(g0, g1, gz4, gz4, gz8, 0);
  }
  for (int i = tid; i < kN * kN; i += 64) mulb[i] = (uint8_t)mul[i];
  if (tid == 0) __builtin_amdgcn_s_wait_tensorcnt(0);
#else
  for (int i = tid; i < kN * kN; i += 64) mulb[i] = (uint8_t)mul[i];
  for (int i = tid; i < kL; i += 64) tok[i] = src[i];
#endif
  __syncthreads();

  int s = tid;
  #pragma unroll 4
  for (int t = 0; t < kL; ++t) {
    int g = tok[t];
    int ns = mulb[g * kN + s];
    s = (tid < kN) ? ns : s;
  }
  perms[(size_t)blk * 64 + tid] = (uint8_t)s;
}

__global__ __launch_bounds__(32) void zscan_compose_wmma(
    const uint8_t* __restrict__ perms, const float* __restrict__ negp,
    float* __restrict__ out)
{
  __shared__ uint32_t vvw[(31 * 64) / 4];
  uint8_t* vv = (uint8_t*)vvw;

  const int lane = threadIdx.x;
  const int b = blockIdx.x;
  const int m_loc = lane & 15;
  const int half  = lane >> 4;

  const uint32_t* src = (const uint32_t*)(perms + (size_t)b * kC * 64);
  for (int i = lane; i < kC * 16; i += 32) vvw[i] = src[i];
  __syncthreads();

  const int off[5] = {0, 16, 24, 28, 30};

  #pragma unroll
  for (int l = 0; l < 4; ++l) {
    const int pairs = kC >> (l + 1);
    for (int i = 0; i < pairs; ++i) {
      const uint8_t* p = vv + (off[l] + 2 * i    ) * 64;
      const uint8_t* q = vv + (off[l] + 2 * i + 1) * 64;
      uint8_t* f = vv + (off[l + 1] + i) * 64;

      v8i Bt[4];
      #pragma unroll
      for (int ni = 0; ni < 4; ++ni) {
        const int pj = p[ni * 16 + m_loc];
        #pragma unroll
        for (int v = 0; v < 8; ++v) {
          const int k0 = ((v & 1) << 2) + (((v >> 1) & 1) << 4) + ((v >> 2) << 5) + (half << 3);
          uint32_t w = 0;
          #pragma unroll
          for (int t2 = 0; t2 < 4; ++t2)
            w |= (pj == (k0 + t2)) ? (1u << (8 * t2)) : 0u;
          Bt[ni][v] = (int)w;
        }
      }

      #pragma unroll
      for (int mi = 0; mi < 4; ++mi) {
        const int mg = mi * 16 + m_loc;
        v8i A;
        #pragma unroll
        for (int v = 0; v < 8; ++v) {
          const int k0 = ((v & 1) << 2) + (((v >> 1) & 1) << 4) +
                         ((v >> 2) << 5) + (half << 3);
          uint32_t w = 0;
          #pragma unroll
          for (int t2 = 0; t2 < 4; ++t2)
            w |= (q[k0 + t2] == mg) ? (1u << (8 * t2)) : 0u;
          A[v] = (int)w;
        }
        #pragma unroll
        for (int ni = 0; ni < 4; ++ni) {
          v8i Cm = {};
          Cm = __builtin_amdgcn_wmma_i32_16x16x64_iu8(false, A, false, Bt[ni],
                                                      Cm, false, false);
          #pragma unroll
          for (int r = 0; r < 8; ++r)
            if (Cm[r] != 0)
              f[ni * 16 + m_loc] = (uint8_t)(mi * 16 + r + 8 * half);
        }
      }
    }
    __syncthreads();
  }

  const int sf = vv[30 * 64 + 0];
  const float neg = negp[0];
  for (int n = lane; n < kN; n += 32)
    out[(size_t)b * kN + n] = (n == sf) ? 0.0f : neg;
}

extern "C" void kernel_launch(void* const* d_in, const int* in_sizes, int n_in,
                              void* d_out, int out_size, void* d_ws, size_t ws_size,
                              hipStream_t stream) {
  (void)in_sizes; (void)n_in; (void)out_size; (void)ws_size;
  const int*   ids = (const int*)d_in[0];
  const int*   mul = (const int*)d_in[1];
  const float* neg = (const float*)d_in[2];

  uint8_t* perms = (uint8_t*)d_ws;

  zscan_chunk_perm<<<kB * kC, 64, 0, stream>>>(ids, mul, perms);
  zscan_compose_wmma<<<kB, 32, 0, stream>>>(perms, neg, (float*)d_out);
}
